// RNN_80479097192492
// MI455X (gfx1250) — matrix operand order repaired and checked by reading, not compiled, not run on hardware
//
#include <hip/hip_runtime.h>
#include <math.h>

// ---------------------------------------------------------------------------
// Seq2seq LSTM forward on gfx1250 (MI455X), f16 WMMA (16x16x32) + f32 acc.
//   B=512, D=128, H=1024, KZ=D+H=1152, 239 enc + 24 dec steps.
// Recurrent GEMM is L2-BW bound -> block tile 128x32(x4 gates), weight slab
// staged to LDS via Tensor Data Mover (triple-buffered, TENSORcnt-paced).
// ---------------------------------------------------------------------------

typedef __attribute__((ext_vector_type(16))) _Float16 v16h;
typedef __attribute__((ext_vector_type(8)))  _Float16 v8h;
typedef __attribute__((ext_vector_type(8)))  float    v8f;
typedef __attribute__((ext_vector_type(4)))  unsigned int u32x4;
typedef __attribute__((ext_vector_type(8)))  int      i32x8;
typedef __attribute__((ext_vector_type(4)))  int      i32x4;

#define BATCH 512
#define DIN   128
#define HID   1024
#define KZ    1152          // DIN + HID
#define G4H   4096          // 4*HID
#define SLEN  240
#define TLEN  24
#define NKC   36            // KZ / 32 k-chunks
#define BPITCH 80           // LDS row pitch: 64B data + 16B TDM pad
#define BSLAB (128 * BPITCH) // 4 gates x 32 rows, padded = 10240 B

// ---- WMMA operand loaders (16-bit A/B, contiguous k order shared by A and B) -------
// A 16x32: lanes 0-15 hold K 0..15 contiguous; lanes 16-31 hold K 16..31
__device__ __forceinline__ v16h ld_a(const _Float16* __restrict__ p) {
    v8h lo = *(const v8h*)(p);
    v8h hi = *(const v8h*)(p + 8);
    return __builtin_shufflevector(lo, hi, 0,1,2,3,4,5,6,7,8,9,10,11,12,13,14,15);
}
// B 32x16: lane n<16 holds col n K 0..15 contiguous; lane n+16 holds K 16..31
__device__ __forceinline__ v16h ld_b16(const char* p) {
    v8h lo = *(const v8h*)(p);
    v8h hi = *(const v8h*)(p + 16);
    return __builtin_shufflevector(lo, hi, 0,1,2,3,4,5,6,7,8,9,10,11,12,13,14,15);
}
__device__ __forceinline__ v8f wmma_f16(v16h a, v16h b, v8f c) {
    return __builtin_amdgcn_wmma_f32_16x16x32_f16(false, a, false, b,
                                                  (short)0, c, false, false);
}
__device__ __forceinline__ float sigmoidf_(float x) {
    return 1.0f / (1.0f + __expf(-x));
}

// ---- TDM: load one k-chunk weight slab (4 gates x 32 rows x 64B) to LDS ----
// 3-D tile: dim0=32 halves (K), dim1=32 rows (stride KZ), dim2=4 gates
// (stride HID*KZ). data_size=2B. pad 4 DW per 16 DW -> 80B LDS pitch.
__device__ __forceinline__ void tdm_load_b(unsigned int lds_off,
                                           const _Float16* gsrc) {
    unsigned long long ga = (unsigned long long)(uintptr_t)gsrc;
    u32x4 g0;
    g0[0] = 1u;                                   // count=1, user mode
    g0[1] = lds_off;                              // lds_addr
    g0[2] = (unsigned int)ga;                     // global_addr[31:0]
    g0[3] = (unsigned int)(ga >> 32) | 0x80000000u; // ga[56:32] | type=2
    i32x8 g1;
    g1[0] = (int)((1u << 16)      // data_size = 2B
                | (1u << 20)      // pad_enable
                | (3u << 22)      // pad_interval: every 16 DWORDs (64B)
                | (3u << 25));    // pad_amount: 4 DWORDs (16B)
    g1[1] = 0;                    // atomic_barrier_addr=0, tensor_dim0[15:0]=0
    g1[2] = 0x10;                 // tensor_dim0 = 1M (no OOB), dim1[15:0]=0
    g1[3] = (32 << 16) | 0x10;    // tile_dim0=32 | tensor_dim1 = 1M
    g1[4] = (4 << 16) | 32;       // tile_dim2=4 | tile_dim1=32
    g1[5] = KZ;                   // tensor_dim0_stride lo
    g1[6] = 0;                    // ts0 hi=0 | tensor_dim1_stride[15:0]=0
    g1[7] = (int)((HID * KZ) >> 16); // tensor_dim1_stride[47:16]
    i32x4 g2; g2[0] = 16; g2[1] = 1; g2[2] = 0; g2[3] = 0;
    i32x4 g3; g3[0] = 0;  g3[1] = 0; g3[2] = 0; g3[3] = 0;
    i32x8 gx;                     // extra operand (clang-23 form): zero-filled
    gx[0] = 0; gx[1] = 0; gx[2] = 0; gx[3] = 0;
    gx[4] = 0; gx[5] = 0; gx[6] = 0; gx[7] = 0;
    __builtin_amdgcn_tensor_load_to_lds(g0, g1, g2, g3, gx, 0);
}

// ---- prep kernels ----------------------------------------------------------
__global__ void prep_wz(const float* __restrict__ Wih, const float* __restrict__ Whh,
                        _Float16* __restrict__ Wz) {
    int idx = blockIdx.x * blockDim.x + threadIdx.x;
    if (idx >= G4H * KZ) return;
    int r = idx / KZ, k = idx % KZ;
    float v = (k < DIN) ? Wih[r * DIN + k] : Whh[(size_t)r * HID + (k - DIN)];
    Wz[idx] = (_Float16)v;
}
__global__ void prep_wout(const float* __restrict__ Wout, _Float16* __restrict__ Wo) {
    int idx = blockIdx.x * blockDim.x + threadIdx.x;
    if (idx >= DIN * HID) return;
    Wo[idx] = (_Float16)Wout[idx];
}
__global__ void prep_bsum(const float* __restrict__ bih, const float* __restrict__ bhh,
                          float* __restrict__ bsum) {
    int idx = blockIdx.x * blockDim.x + threadIdx.x;
    if (idx >= G4H) return;
    bsum[idx] = bih[idx] + bhh[idx];
}
__global__ void init_state(float* __restrict__ cbuf, _Float16* __restrict__ z0) {
    int idx = blockIdx.x * blockDim.x + threadIdx.x;
    if (idx >= BATCH * HID) return;
    cbuf[idx] = 0.0f;
    int b = idx / HID, n = idx % HID;
    z0[(size_t)b * KZ + DIN + n] = (_Float16)0.0f;   // h0 = 0
}
__global__ void load_x(const float* __restrict__ src, _Float16* __restrict__ z, int t) {
    int idx = blockIdx.x * blockDim.x + threadIdx.x;
    if (idx >= BATCH * DIN) return;
    int b = idx / DIN, k = idx % DIN;
    z[(size_t)b * KZ + k] = (_Float16)src[(size_t)b * SLEN * DIN + (size_t)t * DIN + k];
}

// ---- fused LSTM step: gates GEMM (4 gates) + cell update -------------------
// Grid: 128 blocks x 8 waves. Block tile = 128 rows x 32 cols x 4 gates.
// Wave w owns 16 rows; B operands come from the TDM-staged LDS slab and are
// shared by all 8 waves; A chunk is reused across 8 WMMAs per k-step.
__global__ __launch_bounds__(256) void lstm_step(
        const _Float16* __restrict__ Wz,     // [4096][1152] f16
        const float*    __restrict__ bsum,   // [4096]
        const _Float16* __restrict__ zin,    // [512][1152] f16  (x | h_prev)
        _Float16*       __restrict__ zout,   // h_new -> zout[:, DIN:]
        float*          __restrict__ cbuf) { // [512][1024] f32
    __shared__ __align__(16) char smem[3][BSLAB];   // 30 KB, triple buffer

    const int tid  = threadIdx.x;
    const int lane = tid & 31;
    const int wave = tid >> 5;            // 0..7 -> 16-row M-subtile
    const int mblk = blockIdx.x >> 5;     // 0..3  -> rows mblk*128
    const int nblk = blockIdx.x & 31;     // 0..31 -> cols nblk*32
    const int lsel = lane & 15;
    const int hiH  = lane >> 4;
    const int ka   = hiH ? 16 : 0;

    const _Float16* arow  = zin + (size_t)(mblk * 128 + wave * 16 + lsel) * KZ;
    const _Float16* bbase = Wz + (size_t)(nblk * 32) * KZ;   // gates via TDM dim2
    const unsigned int sbase = (unsigned int)(uintptr_t)(&smem[0][0]);
    // lane-fixed LDS read base; operand (g,ns) lives at +(g*2+ns)*1280
    const char* sb_lane = &smem[0][0] + lsel * BPITCH + hiH * 32;

    v8f acc[8] = {};   // flat index i = g*2 + ns

    if (wave == 0) {      // prologue: 2 chunks in flight
        tdm_load_b(sbase + 0 * BSLAB, bbase + 0);
        tdm_load_b(sbase + 1 * BSLAB, bbase + 32);
    }

    unsigned cur = 0;            // LDS byte offset of chunk kc's slab
    unsigned nxt = 2 * BSLAB;    // LDS byte offset for chunk kc+2
    #define LSTM_CHUNK(kc_)                                                   \
    {                                                                         \
        v16h va = ld_a(arow + (kc_) * 32 + ka);                               \
        const char* p = sb_lane + cur;                                        \
        v16h vb0 = ld_b16(p);                                                 \
        v16h vb1 = ld_b16(p + 1 * 16 * BPITCH);                               \
        v16h vb2 = ld_b16(p + 2 * 16 * BPITCH);                               \
        v16h vb3 = ld_b16(p + 3 * 16 * BPITCH);                               \
        v16h vb4 = ld_b16(p + 4 * 16 * BPITCH);                               \
        v16h vb5 = ld_b16(p + 5 * 16 * BPITCH);                               \
        v16h vb6 = ld_b16(p + 6 * 16 * BPITCH);                               \
        v16h vb7 = ld_b16(p + 7 * 16 * BPITCH);                               \
        acc[0] = wmma_f16(va, vb0, acc[0]);                                   \
        acc[1] = wmma_f16(va, vb1, acc[1]);                                   \
        acc[2] = wmma_f16(va, vb2, acc[2]);                                   \
        acc[3] = wmma_f16(va, vb3, acc[3]);                                   \
        acc[4] = wmma_f16(va, vb4, acc[4]);                                   \
        acc[5] = wmma_f16(va, vb5, acc[5]);                                   \
        acc[6] = wmma_f16(va, vb6, acc[6]);                                   \
        acc[7] = wmma_f16(va, vb7, acc[7]);                                   \
    }

    // steady state: always wait<=1 (retires chunk kc; kc+1 stays in flight)
    for (int kc = 0; kc < NKC - 1; ++kc) {
        if (wave == 0) __builtin_amdgcn_s_wait_tensorcnt((short)1);
        __syncthreads();  // slab kc visible; all waves done reading slab kc-1
        if (wave == 0 && kc + 2 < NKC)
            tdm_load_b(sbase + nxt, bbase + (kc + 2) * 32);
        LSTM_CHUNK(kc)
        cur += BSLAB; if (cur == 3 * BSLAB) cur = 0;
        nxt += BSLAB; if (nxt == 3 * BSLAB) nxt = 0;
    }
    // final chunk: drain TENSORcnt fully
    if (wave == 0) __builtin_amdgcn_s_wait_tensorcnt((short)0);
    __syncthreads();
    LSTM_CHUNK(NKC - 1)
    #undef LSTM_CHUNK

    // Epilogue: D layout -> lane l<16: (M=r, N=l); l>=16: (M=8+r, N=l-16)
    #pragma unroll
    for (int ns = 0; ns < 2; ++ns) {
        const int n  = nblk * 32 + ns * 16 + lsel;
        const float bi = bsum[n];
        const float bf = bsum[HID + n];
        const float bg = bsum[2 * HID + n];
        const float bo = bsum[3 * HID + n];
        const int mbase = mblk * 128 + wave * 16 + hiH * 8;
        #pragma unroll
        for (int r = 0; r < 8; ++r) {
            const int m = mbase + r;
            float ig = sigmoidf_(acc[0 * 2 + ns][r] + bi);
            float fg = sigmoidf_(acc[1 * 2 + ns][r] + bf);
            float gg = tanhf   (acc[2 * 2 + ns][r] + bg);
            float og = sigmoidf_(acc[3 * 2 + ns][r] + bo);
            size_t ci = (size_t)m * HID + n;
            float c = fg * cbuf[ci] + ig * gg;
            cbuf[ci] = c;
            zout[(size_t)m * KZ + DIN + n] = (_Float16)(og * tanhf(c));
        }
    }
}

// ---- decoder output projection: out = h @ W_out^T + b_out ------------------
__global__ __launch_bounds__(256) void proj_step(
        const _Float16* __restrict__ Wo,     // [128][1024] f16
        const float*    __restrict__ bout,   // [128]
        _Float16*       __restrict__ z,      // read h at +DIN, write x at [0,DIN)
        float*          __restrict__ out, int t) {
    const int lane  = threadIdx.x & 31;
    const int wid   = (blockIdx.x * blockDim.x + threadIdx.x) >> 5;  // 0..127
    const int mtile = wid >> 3;          // 0..15
    const int ntile = wid & 7;           // 0..7
    const int lsel  = lane & 15;
    const int hiH   = lane >> 4;
    const int ka    = hiH ? 16 : 0;
    const int kb    = hiH ? 16 : 0;

    const _Float16* a0 = z + (size_t)(mtile * 32 + lsel) * KZ + DIN;
    const _Float16* a1 = a0 + (size_t)16 * KZ;
    const _Float16* br = Wo + (size_t)(ntile * 16 + lsel) * HID;

    v8f acc[2] = {};
    #pragma unroll 4
    for (int k = 0; k < HID; k += 32) {
        v16h va0 = ld_a(a0 + k + ka);
        v16h va1 = ld_a(a1 + k + ka);
        v8h blo = *(const v8h*)(br + k + kb);
        v8h bhi = *(const v8h*)(br + k + kb + 8);
        v16h vb = __builtin_shufflevector(blo, bhi,
                    0,1,2,3,4,5,6,7,8,9,10,11,12,13,14,15);
        acc[0] = wmma_f16(va0, vb, acc[0]);
        acc[1] = wmma_f16(va1, vb, acc[1]);
    }

    const int n = ntile * 16 + lsel;
    const float bb = bout[n];
    #pragma unroll
    for (int sub = 0; sub < 2; ++sub) {
        const int mbase = mtile * 32 + sub * 16 + (hiH ? 8 : 0);
        #pragma unroll
        for (int r = 0; r < 8; ++r) {
            const int m = mbase + r;
            float v = acc[sub][r] + bb;
            out[((size_t)m * TLEN + t) * DIN + n] = v;      // [B, 24, 128]
            z[(size_t)m * KZ + n] = (_Float16)v;            // next decoder input
        }
    }
}

// ---------------------------------------------------------------------------
extern "C" void kernel_launch(void* const* d_in, const int* in_sizes, int n_in,
                              void* d_out, int out_size, void* d_ws, size_t ws_size,
                              hipStream_t stream) {
    const float* src  = (const float*)d_in[0];
    // d_in[1] = tgt: only its length (24) matters; compile-time constant.
    const float* Wih  = (const float*)d_in[2];
    const float* Whh  = (const float*)d_in[3];
    const float* bih  = (const float*)d_in[4];
    const float* bhh  = (const float*)d_in[5];
    const float* Wout = (const float*)d_in[6];
    const float* bout = (const float*)d_in[7];
    float* out = (float*)d_out;

    char* ws = (char*)d_ws;
    size_t off = 0;
    _Float16* Wz   = (_Float16*)(ws + off); off += (size_t)G4H * KZ * 2;      // 9.0 MB
    _Float16* Wo   = (_Float16*)(ws + off); off += (size_t)DIN * HID * 2;     // 256 KB
    float*    bsum = (float*)   (ws + off); off += (size_t)G4H * 4;           // 16 KB
    _Float16* zb[2];
    zb[0] = (_Float16*)(ws + off); off += (size_t)BATCH * KZ * 2;             // 1.13 MB
    zb[1] = (_Float16*)(ws + off); off += (size_t)BATCH * KZ * 2;             // 1.13 MB
    float*    cbuf = (float*)   (ws + off); off += (size_t)BATCH * HID * 4;   // 2.0 MB

    prep_wz  <<<(G4H * KZ + 255) / 256, 256, 0, stream>>>(Wih, Whh, Wz);
    prep_wout<<<(DIN * HID + 255) / 256, 256, 0, stream>>>(Wout, Wo);
    prep_bsum<<<(G4H + 255) / 256,        256, 0, stream>>>(bih, bhh, bsum);
    init_state<<<(BATCH * HID + 255) / 256, 256, 0, stream>>>(cbuf, zb[0]);

    const dim3 stepGrid(128), projGrid(16), blk(256);
    const int xGrid = (BATCH * DIN + 255) / 256;

    // -- encoder: 239 steps over src[0..238], h ping-pongs between z-buffers --
    int p = 0;
    for (int t = 0; t < SLEN - 1; ++t) {
        load_x   <<<xGrid, blk, 0, stream>>>(src, zb[p], t);
        lstm_step<<<stepGrid, blk, 0, stream>>>(Wz, bsum, zb[p], zb[1 - p], cbuf);
        p ^= 1;
    }

    // -- decoder: seed with last src frame, then autoregress for 24 steps --
    load_x<<<xGrid, blk, 0, stream>>>(src, zb[p], SLEN - 1);
    for (int t = 0; t < TLEN; ++t) {
        lstm_step<<<stepGrid, blk, 0, stream>>>(Wz, bsum, zb[p], zb[1 - p], cbuf);
        proj_step<<<projGrid, blk, 0, stream>>>(Wo, bout, zb[1 - p], out, t);
        p ^= 1;
    }
}
